// Attention_41532333753073
// MI455X (gfx1250) — matrix operand order repaired and checked by reading, not compiled, not run on hardware
//
#include <hip/hip_runtime.h>
#include <hip/hip_bf16.h>
#include <math.h>

// ---------------------------------------------------------------------------
// Types for CDNA5 WMMA (wave32). v16bf = A/B fragment (8 VGPRs), v8f = C/D.
// ---------------------------------------------------------------------------
typedef __bf16 bf16_t;
typedef __attribute__((ext_vector_type(16))) __bf16 v16bf;
typedef __attribute__((ext_vector_type(8)))  float  v8f;
typedef __attribute__((ext_vector_type(4)))  int    v4i;

union Frag16 {            // 32-byte A/B fragment, loadable as two uint4
    v16bf v;
    uint4 q[2];
};

static __device__ __forceinline__ v8f wmma_bf16(v16bf a, v16bf b, v8f c) {
    return __builtin_amdgcn_wmma_f32_16x16x32_bf16(
        /*neg_a=*/false, a, /*neg_b=*/false, b,
        /*c_mod=*/(short)0, c, /*reuse_a=*/false, /*reuse_b=*/false);
}

// ---------------------------------------------------------------------------
// CDNA5 async global->LDS staging (ASYNCcnt). Guarded: falls back to a sync
// copy if the toolchain lacks the builtins. Signature (probe-derived):
//   __builtin_amdgcn_global_load_async_to_lds_b128(v4i AS1*, v4i AS3*,
//                                                  imm offset, imm cpol)
// ---------------------------------------------------------------------------
#if defined(__has_builtin)
#if __has_builtin(__builtin_amdgcn_global_load_async_to_lds_b128)
#define HAVE_ASYNC_LDS 1
#endif
#endif
#ifndef HAVE_ASYNC_LDS
#define HAVE_ASYNC_LDS 0
#endif

#define GLOBAL_AS __attribute__((address_space(1)))
#define LDS_AS    __attribute__((address_space(3)))

static __device__ __forceinline__ void async_copy_b128(const bf16_t* g,
                                                       bf16_t* l) {
#if HAVE_ASYNC_LDS
    __builtin_amdgcn_global_load_async_to_lds_b128(
        (GLOBAL_AS v4i*)(g), (LDS_AS v4i*)(l), /*offset=*/0, /*cpol=*/0);
#else
    *(uint4*)l = *(const uint4*)g;
#endif
}

static __device__ __forceinline__ void wait_async_all() {
#if HAVE_ASYNC_LDS
#if __has_builtin(__builtin_amdgcn_s_wait_asynccnt)
    __builtin_amdgcn_s_wait_asynccnt(0);
#else
    asm volatile("s_wait_asynccnt 0x0" ::: "memory");
#endif
#endif
}

// ---------------------------------------------------------------------------
// fp32 -> bf16 converter (grid-stride)
// ---------------------------------------------------------------------------
__global__ void cvt_f32_bf16_kernel(const float* __restrict__ in,
                                    bf16_t* __restrict__ out, size_t n) {
    size_t i = (size_t)blockIdx.x * blockDim.x + threadIdx.x;
    size_t stride = (size_t)gridDim.x * blockDim.x;
    for (; i < n; i += stride) out[i] = (bf16_t)in[i];
}

// ---------------------------------------------------------------------------
// fp32 [rows][cols] -> bf16 transposed [cols][rows], 32x32 LDS tiles.
// Lets the GEMM B-operand be staged with straight (async) row copies.
// ---------------------------------------------------------------------------
__global__ __launch_bounds__(256) void transpose_cvt_kernel(
    const float* __restrict__ in, bf16_t* __restrict__ out,
    int rows, int cols)
{
    __shared__ float tile[32][33];
    const int tx = threadIdx.x & 31;
    const int ty = threadIdx.x >> 5;          // 0..7
    const int r0 = blockIdx.y * 32;
    const int c0 = blockIdx.x * 32;
#pragma unroll
    for (int i = 0; i < 32; i += 8)
        tile[ty + i][tx] = in[(size_t)(r0 + ty + i) * cols + c0 + tx];
    __syncthreads();
#pragma unroll
    for (int i = 0; i < 32; i += 8)
        out[(size_t)(c0 + ty + i) * rows + r0 + tx] = (bf16_t)tile[tx][ty + i];
}

// ---------------------------------------------------------------------------
// Tiled bf16 WMMA GEMM: C[f32, MxN] = A[bf16, MxK, row-major] *
// Bt[bf16, NxK, row-major]  (i.e. B pre-transposed).
// Block: 256 thr (8 waves). Tile 128x128, BK=32, double-buffered async LDS
// staging. Wave grid 4(M) x 2(N); each wave computes 32x64 = 2x4 WMMA tiles.
// M%128==0, N%128==0, K%32==0 assumed.
// ---------------------------------------------------------------------------
#define GBM 128
#define GBN 128
#define GBK 32

__global__ __launch_bounds__(256) void gemm_bf16_wmma_kernel(
    const bf16_t* __restrict__ A, const bf16_t* __restrict__ Bt,
    float* __restrict__ C, int M, int N, int K)
{
    __shared__ __align__(16) bf16_t lA[2][GBM][GBK];   // 16 KB
    __shared__ __align__(16) bf16_t lBt[2][GBN][GBK];  // 16 KB

    const int tid  = threadIdx.x;
    const int lane = tid & 31;
    const int wave = tid >> 5;
    const int wm   = wave >> 1;       // 0..3  -> M offset wm*32
    const int wn   = wave & 1;        // 0..1  -> N offset wn*64
    const int ln   = lane & 15;
    const int hl   = lane >> 4;       // half-wave select

    const int bm = blockIdx.y * GBM;
    const int bn = blockIdx.x * GBN;

    // Per-thread staging slots: 2 x b128 per matrix per tile.
    const int sr = tid >> 1;              // row 0..127
    const int sc = (tid & 1) * 16;        // col 0 or 16 (8 elems each half)

    auto stage = [&](int buf, int k0) {
#pragma unroll
        for (int i = 0; i < 2; ++i) {
            int c = sc + i * 8;
            async_copy_b128(&A[(size_t)(bm + sr) * K + k0 + c],
                            &lA[buf][sr][c]);
            async_copy_b128(&Bt[(size_t)(bn + sr) * K + k0 + c],
                            &lBt[buf][sr][c]);
        }
    };

    v8f acc[2][4];
#pragma unroll
    for (int mt = 0; mt < 2; ++mt)
#pragma unroll
        for (int nt = 0; nt < 4; ++nt) acc[mt][nt] = (v8f){};

    // Prologue: fill buffer 0.
    stage(0, 0);
    wait_async_all();
    __syncthreads();

    int cur = 0;
    for (int k0 = 0; k0 < K; k0 += GBK) {
        // Kick off next tile into the other buffer while we compute.
        if (k0 + GBK < K) stage(cur ^ 1, k0 + GBK);

        // ---- A fragments: lane row = wm*32 + mt*16 + ln;
        //      elems 0..7 -> k = 8*hl.., elems 8..15 -> k = 16+8*hl.. ----
        Frag16 af[2];
#pragma unroll
        for (int mt = 0; mt < 2; ++mt) {
            const bf16_t* row = &lA[cur][wm * 32 + mt * 16 + ln][0];
            af[mt].q[0] = *(const uint4*)&row[8 * hl];
            af[mt].q[1] = *(const uint4*)&row[16 + 8 * hl];
        }
        // ---- B fragments: lane col = wn*64 + nt*16 + ln;
        //      elems 0..7 -> k = 8*hl.., elems 8..15 -> k = 16+8*hl.. ----
        Frag16 bf[4];
#pragma unroll
        for (int nt = 0; nt < 4; ++nt) {
            const bf16_t* col = &lBt[cur][wn * 64 + nt * 16 + ln][0];
            bf[nt].q[0] = *(const uint4*)&col[8 * hl];
            bf[nt].q[1] = *(const uint4*)&col[16 + 8 * hl];
        }
#pragma unroll
        for (int mt = 0; mt < 2; ++mt)
#pragma unroll
            for (int nt = 0; nt < 4; ++nt)
                acc[mt][nt] = wmma_bf16(af[mt].v, bf[nt].v, acc[mt][nt]);

        // Next buffer must be fully written (all waves) before next iter.
        wait_async_all();
        __syncthreads();
        cur ^= 1;
    }

    // ---- store C: elem j -> row m = j + 8*hl, col n = ln ----
#pragma unroll
    for (int mt = 0; mt < 2; ++mt)
#pragma unroll
        for (int nt = 0; nt < 4; ++nt)
#pragma unroll
            for (int j = 0; j < 8; ++j) {
                int row = bm + wm * 32 + mt * 16 + j + 8 * hl;
                int col = bn + wn * 64 + nt * 16 + ln;
                C[(size_t)row * N + col] = acc[mt][nt][j];
            }
}

// ---------------------------------------------------------------------------
// RoPE on q,k + v = lambda1*v1 + lambda2*v, fp32 -> bf16, permute
// [B][T][H*64] -> [B*H][T][64]. q pre-scaled by 1/sqrt(Dh).
// One thread per (b,t,h,i), i = rotary pair 0..31.
// ---------------------------------------------------------------------------
__global__ void rope_mix_kernel(
    const float* __restrict__ Qf, const float* __restrict__ Kf,
    const float* __restrict__ Vf, const float* __restrict__ v1,
    const float* __restrict__ lam1p, const float* __restrict__ lam2p,
    bf16_t* __restrict__ Qb, bf16_t* __restrict__ Kb, bf16_t* __restrict__ Vb,
    int T, float qscale)
{
    size_t idx = (size_t)blockIdx.x * blockDim.x + threadIdx.x;
    const int    i  = (int)(idx & 31);
    const int    h  = (int)((idx >> 5) & 31);
    const size_t bt = idx >> 10;                 // b*T + t
    const int    t  = (int)(bt % (size_t)T);
    const int    b  = (int)(bt / (size_t)T);

    const float inv_freq = __powf(10000.0f, -(float)(2 * i) / 64.0f);
    const float ang = (float)t * inv_freq;
    const float c = __cosf(ang), s = __sinf(ang);

    const size_t src = bt * 2048 + (size_t)h * 64;
    const float q1 = Qf[src + i], q2 = Qf[src + i + 32];
    const float k1 = Kf[src + i], k2 = Kf[src + i + 32];
    const float l1 = lam1p[0],    l2 = lam2p[0];
    const float va = l1 * v1[src + i]      + l2 * Vf[src + i];
    const float vb = l1 * v1[src + i + 32] + l2 * Vf[src + i + 32];

    const size_t dst = (((size_t)(b * 32 + h)) * T + t) * 64;
    Qb[dst + i]      = (bf16_t)(qscale * (q1 * c - q2 * s));
    Qb[dst + i + 32] = (bf16_t)(qscale * (q2 * c + q1 * s));
    Kb[dst + i]      = (bf16_t)(k1 * c - k2 * s);
    Kb[dst + i + 32] = (bf16_t)(k2 * c + k1 * s);
    Vb[dst + i]      = (bf16_t)va;
    Vb[dst + i + 32] = (bf16_t)vb;
}

// ---------------------------------------------------------------------------
// Flash attention (causal), WMMA bf16. Layouts: Q/K/V/O = [B*H][T][64].
// Block = 128 threads (4 waves); each block does 64 query rows, each wave 16.
// K-tiles of 64 keys staged in LDS (K async, V sync-transposed).
// ---------------------------------------------------------------------------
__global__ __launch_bounds__(128) void flash_attn_kernel(
    const bf16_t* __restrict__ Qb, const bf16_t* __restrict__ Kb,
    const bf16_t* __restrict__ Vb, float* __restrict__ Of, int T)
{
    constexpr int DH = 64;
    const int bh   = blockIdx.y;
    const int q0   = blockIdx.x * 64;
    const int tid  = threadIdx.x;
    const int wave = tid >> 5;
    const int lane = tid & 31;
    const int ln   = lane & 15;
    const int hl   = lane >> 4;

    __shared__ __align__(16) bf16_t lK[64][DH];      // [key][dh]   8 KB
    __shared__ __align__(16) bf16_t lVt[DH][64];     // [dh][key]   8 KB
    __shared__ __align__(16) bf16_t lP[4][16][64];   // per-wave P  8 KB

    const bf16_t* Qh = Qb + (size_t)bh * T * DH;
    const bf16_t* Kh = Kb + (size_t)bh * T * DH;
    const bf16_t* Vh = Vb + (size_t)bh * T * DH;

    // Q fragments (A layout), reused across all key tiles. Lane row:
    const int qrow = q0 + wave * 16 + ln;
    Frag16 aq[2];
#pragma unroll
    for (int kk = 0; kk < 2; ++kk) {
        const bf16_t* r = &Qh[(size_t)qrow * DH + kk * 32];
        aq[kk].q[0] = *(const uint4*)&r[8 * hl];
        aq[kk].q[1] = *(const uint4*)&r[16 + 8 * hl];
    }

    v8f oacc[4];
#pragma unroll
    for (int nt = 0; nt < 4; ++nt) oacc[nt] = (v8f){};
    float mrow[8], lrow[8];
#pragma unroll
    for (int j = 0; j < 8; ++j) { mrow[j] = -3.0e38f; lrow[j] = 0.0f; }

    const int ktiles = (q0 + 64) / 64;     // causal: up to & incl. diagonal
    for (int kt = 0; kt < ktiles; ++kt) {
        const int k0 = kt * 64;
        __syncthreads();
        // stage K (async) and transposed V (sync): 4 x b128 per thread each
#pragma unroll
        for (int i = 0; i < 4; ++i) {
            int v = tid + 128 * i;          // 512 vectors
            int r = v >> 3;                 // key 0..63
            int c = (v & 7) * 8;            // dh 0..56
            async_copy_b128(&Kh[(size_t)(k0 + r) * DH + c], &lK[r][c]);
            __align__(16) bf16_t tmp[8];
            *(uint4*)tmp = *(const uint4*)&Vh[(size_t)(k0 + r) * DH + c];
#pragma unroll
            for (int e = 0; e < 8; ++e) lVt[c + e][r] = tmp[e];
        }
        wait_async_all();
        __syncthreads();

        // ---- S = Q * K^T : 4 key sub-tiles of 16, K-dim = DH = 2 WMMAs ----
        v8f s[4];
#pragma unroll
        for (int nt = 0; nt < 4; ++nt) {
            v8f a = (v8f){};
#pragma unroll
            for (int kk = 0; kk < 2; ++kk) {
                Frag16 bk;                  // B elems: dh = kk*32 + table order (8*hl / 16+8*hl)
                const bf16_t* row = &lK[nt * 16 + ln][kk * 32];
                bk.q[0] = *(const uint4*)&row[8 * hl];
                bk.q[1] = *(const uint4*)&row[16 + 8 * hl];
                a = wmma_bf16(aq[kk].v, bk.v, a);
            }
            s[nt] = a;
        }

        // ---- causal mask + online softmax (row = j + 8*hl within wave) ----
#pragma unroll
        for (int j = 0; j < 8; ++j) {
            const int qr = q0 + wave * 16 + j + 8 * hl;
            float rmax = -3.0e38f;
#pragma unroll
            for (int nt = 0; nt < 4; ++nt) {
                const int kc = k0 + nt * 16 + ln;
                float sv = s[nt][j];
                if (kc > qr) sv = -3.0e38f;
                s[nt][j] = sv;
                rmax = fmaxf(rmax, sv);
            }
#pragma unroll
            for (int off = 1; off < 16; off <<= 1)
                rmax = fmaxf(rmax, __shfl_xor(rmax, off, 32));
            const float mnew = fmaxf(mrow[j], rmax);
            const float corr = __expf(mrow[j] - mnew);
            float rsum = 0.0f;
#pragma unroll
            for (int nt = 0; nt < 4; ++nt) {
                float p = __expf(s[nt][j] - mnew);
                s[nt][j] = p;
                rsum += p;
            }
#pragma unroll
            for (int off = 1; off < 16; off <<= 1)
                rsum += __shfl_xor(rsum, off, 32);
            lrow[j] = lrow[j] * corr + rsum;
            mrow[j] = mnew;
#pragma unroll
            for (int nt = 0; nt < 4; ++nt)
                oacc[nt][j] = oacc[nt][j] * corr;
            // spill P (bf16) to per-wave LDS scratch for C->A re-layout
#pragma unroll
            for (int nt = 0; nt < 4; ++nt)
                lP[wave][j + 8 * hl][nt * 16 + ln] = (bf16_t)s[nt][j];
        }
        __syncthreads();

        // ---- O += P * V : K-dim = 64 keys = 2 WMMAs, 4 dh sub-tiles ----
#pragma unroll
        for (int kk = 0; kk < 2; ++kk) {
            Frag16 ap;                      // A layout from lP
            const bf16_t* pr = &lP[wave][ln][kk * 32];
            ap.q[0] = *(const uint4*)&pr[8 * hl];
            ap.q[1] = *(const uint4*)&pr[16 + 8 * hl];
#pragma unroll
            for (int nt = 0; nt < 4; ++nt) {
                Frag16 bv;                  // B elems: key = kk*32 + table order (8*hl / 16+8*hl)
                const bf16_t* vr = &lVt[nt * 16 + ln][kk * 32];
                bv.q[0] = *(const uint4*)&vr[8 * hl];
                bv.q[1] = *(const uint4*)&vr[16 + 8 * hl];
                oacc[nt] = wmma_bf16(ap.v, bv.v, oacc[nt]);
            }
        }
    }

    // ---- epilogue: normalize, store fp32 O in [bh][t][dh] ----
#pragma unroll
    for (int j = 0; j < 8; ++j) {
        const int qr = q0 + wave * 16 + j + 8 * hl;
        const float inv = 1.0f / lrow[j];
#pragma unroll
        for (int nt = 0; nt < 4; ++nt)
            Of[((size_t)bh * T + qr) * DH + nt * 16 + ln] =
                oacc[nt][j] * inv;
    }
}

// ---------------------------------------------------------------------------
// [B*H][T][64] f32 -> [B*T][H*64] bf16 permute (dest-linear indexing)
// ---------------------------------------------------------------------------
__global__ void permute_o_kernel(const float* __restrict__ Of,
                                 bf16_t* __restrict__ Ob, int T) {
    size_t idx = (size_t)blockIdx.x * blockDim.x + threadIdx.x;
    const int    col = (int)(idx & 2047);
    const int    d   = col & 63;
    const int    h   = col >> 6;
    const size_t bt  = idx >> 11;
    const int    t   = (int)(bt % (size_t)T);
    const int    b   = (int)(bt / (size_t)T);
    Ob[idx] = (bf16_t)Of[(((size_t)(b * 32 + h)) * T + t) * 64 + d];
}

// ---------------------------------------------------------------------------
// Host-side orchestration
// ---------------------------------------------------------------------------
extern "C" void kernel_launch(void* const* d_in, const int* in_sizes, int n_in,
                              void* d_out, int out_size, void* d_ws,
                              size_t ws_size, hipStream_t stream) {
    const float* hid  = (const float*)d_in[0];
    const float* v1   = (const float*)d_in[1];
    const float* lam1 = (const float*)d_in[2];
    const float* Wq   = (const float*)d_in[3];
    const float* Wk   = (const float*)d_in[4];
    const float* Wv   = (const float*)d_in[5];
    const float* Wo   = (const float*)d_in[6];
    const float* lam2 = (const float*)d_in[7];
    float* out = (float*)d_out;

    const int Dm = 2048, H = 32;
    const int BT = in_sizes[0] / Dm;     // B*T rows (4096)
    const int B  = 2;                    // per reference setup
    const int T  = BT / B;

    // ---- workspace carve-out (256B aligned) ----
    char* p = (char*)d_ws;
    auto carve = [&](size_t bytes) {
        char* r = p;
        p += (bytes + 255) & ~(size_t)255;
        return r;
    };
    const size_t nX = (size_t)BT * Dm;        // 8M elems
    const size_t nW = (size_t)Dm * Dm;        // 4M elems
    bf16_t* Xb  = (bf16_t*)carve(nX * 2);
    bf16_t* WqT = (bf16_t*)carve(nW * 2);     // transposed bf16 weights [n][k]
    bf16_t* WkT = (bf16_t*)carve(nW * 2);
    bf16_t* WvT = (bf16_t*)carve(nW * 2);
    bf16_t* WoT = (bf16_t*)carve(nW * 2);
    float*  Qf  = (float*) carve(nX * 4);
    float*  Kf  = (float*) carve(nX * 4);
    float*  Vf  = (float*) carve(nX * 4);
    bf16_t* Qb  = (bf16_t*)carve(nX * 2);
    bf16_t* Kb  = (bf16_t*)carve(nX * 2);
    bf16_t* Vb  = (bf16_t*)carve(nX * 2);
    float*  Of  = (float*) carve(nX * 4);
    bf16_t* Ob  = (bf16_t*)carve(nX * 2);

    // ---- 1) fp32 -> bf16 conversions (weights transposed) ----
    cvt_f32_bf16_kernel<<<4096, 256, 0, stream>>>(hid, Xb, nX);
    dim3 tgrid(Dm / 32, Dm / 32);
    transpose_cvt_kernel<<<tgrid, 256, 0, stream>>>(Wq, WqT, Dm, Dm);
    transpose_cvt_kernel<<<tgrid, 256, 0, stream>>>(Wk, WkT, Dm, Dm);
    transpose_cvt_kernel<<<tgrid, 256, 0, stream>>>(Wv, WvT, Dm, Dm);
    transpose_cvt_kernel<<<tgrid, 256, 0, stream>>>(Wo, WoT, Dm, Dm);

    // ---- 2) Q/K/V projections (WMMA GEMM, async double-buffered) ----
    dim3 ggrid(Dm / GBN, BT / GBM);
    gemm_bf16_wmma_kernel<<<ggrid, 256, 0, stream>>>(Xb, WqT, Qf, BT, Dm, Dm);
    gemm_bf16_wmma_kernel<<<ggrid, 256, 0, stream>>>(Xb, WkT, Kf, BT, Dm, Dm);
    gemm_bf16_wmma_kernel<<<ggrid, 256, 0, stream>>>(Xb, WvT, Vf, BT, Dm, Dm);

    // ---- 3) RoPE + v-mix + permute to [B*H][T][64] bf16 ----
    const float qscale = 0.125f;         // 1/sqrt(64)
    rope_mix_kernel<<<(unsigned)((size_t)BT * H * 32 / 256), 256, 0, stream>>>(
        Qf, Kf, Vf, v1, lam1, lam2, Qb, Kb, Vb, T, qscale);

    // ---- 4) causal flash attention ----
    dim3 agrid(T / 64, B * H);
    flash_attn_kernel<<<agrid, 128, 0, stream>>>(Qb, Kb, Vb, Of, T);

    // ---- 5) permute O back, final projection into d_out ----
    permute_o_kernel<<<(unsigned)(nX / 256), 256, 0, stream>>>(Of, Ob, T);
    gemm_bf16_wmma_kernel<<<ggrid, 256, 0, stream>>>(Ob, WoT, out, BT, Dm, Dm);
}
